// SWMLP_40518721471006
// MI455X (gfx1250) — matrix operand order repaired and checked by reading, not compiled, not run on hardware
//
#include <hip/hip_runtime.h>
#include <hip/hip_bf16.h>

typedef __attribute__((ext_vector_type(8)))  __bf16 v8bf;
typedef __attribute__((ext_vector_type(16))) __bf16 v16bf;
typedef __attribute__((ext_vector_type(8)))  float  v8f;

// ---------------- ws (scratch) layout: packed bf16 weights ----------------
static constexpr int N_W1T   = 3 * 512 * 256;   // W1 transposed: [br][n][k]
static constexpr int N_W2T   = 3 * 256 * 256;   // W2 transposed: [br][n][k]
static constexpr int OFF_W1T = 0;
static constexpr int OFF_W2T = OFF_W1T + N_W1T;
static constexpr int OFF_WSF = OFF_W2T + N_W2T; // Ws_full  [64][64]
static constexpr int OFF_WSH = OFF_WSF + 64*64; // Ws_half  [32][32]
static constexpr int OFF_WSQ = OFF_WSH + 32*32; // Ws_quat  [16][32] (K zero-padded)
static constexpr int TOTAL_PACK = OFF_WSQ + 16*32;

static __device__ __forceinline__ v8f wmma_bf16(v16bf a, v16bf b, v8f c) {
  return __builtin_amdgcn_wmma_f32_16x16x32_bf16(false, a, false, b, (short)0, c, false, false);
}

// A-operand (M=16 x K=32): lane holds row (lane&15); chunks {8*hi.. , 16+8*hi..}
static __device__ __forceinline__ v16bf load_a_frag(const __bf16* base, int rowStride,
                                                    int row, int k0, int hi) {
  const __bf16* p = base + (size_t)row * rowStride + k0 + 8 * hi;
  v8bf lo = *(const v8bf*)p;
  v8bf hh = *(const v8bf*)(p + 16);
  v16bf r;
#pragma unroll
  for (int i = 0; i < 8; ++i) { r[i] = lo[i]; r[8 + i] = hh[i]; }
  return r;
}

// B-operand (K=32 x N=16) from N-major (transposed) storage: lane = column,
// K chunks {8*hi.. , 16+8*hi..}
static __device__ __forceinline__ v16bf load_b_frag(const __bf16* rowptr, int k0, int hi) {
  const __bf16* p = rowptr + k0 + 8 * hi;
  v8bf lo = *(const v8bf*)p;
  v8bf hh = *(const v8bf*)(p + 16);
  v16bf r;
#pragma unroll
  for (int i = 0; i < 8; ++i) { r[i] = lo[i]; r[8 + i] = hh[i]; }
  return r;
}

static __device__ __forceinline__ float gelu_f(float z) {
  return 0.5f * z * (1.0f + erff(z * 0.70710678118654752f));
}

// -------- prep: f32 -> bf16, transpose W1/W2 to N-major, pack Ws ----------
__global__ __launch_bounds__(256) void swmlp_prep(const float* __restrict__ W1,
                                                  const float* __restrict__ W2,
                                                  const float* __restrict__ WsF,
                                                  const float* __restrict__ WsH,
                                                  const float* __restrict__ WsQ,
                                                  __bf16* __restrict__ ws) {
  int i = blockIdx.x * 256 + threadIdx.x;
  if (i < N_W1T) {
    int br = i / 131072, r = i % 131072, n = r / 256, k = r % 256;
    ws[OFF_W1T + i] = (__bf16)W1[(size_t)br * 131072 + (size_t)k * 512 + n];
  } else if (i < N_W1T + N_W2T) {
    int j = i - N_W1T;
    int br = j / 65536, r = j % 65536, n = r / 256, k = r % 256;
    ws[OFF_W2T + j] = (__bf16)W2[(size_t)br * 65536 + (size_t)k * 256 + n];
  } else if (i < OFF_WSH) {
    int j = i - OFF_WSF;
    ws[OFF_WSF + j] = (__bf16)WsF[j];
  } else if (i < OFF_WSQ) {
    int j = i - OFF_WSH;
    ws[OFF_WSH + j] = (__bf16)WsH[j];
  } else if (i < TOTAL_PACK) {
    int j = i - OFF_WSQ, m = j / 32, kk = j % 32;
    ws[OFF_WSQ + j] = (kk < 16) ? (__bf16)WsQ[m * 16 + kk] : (__bf16)0.f;
  }
}

// ---------------- main fused gMLP kernel: one window per block ----------------
// MODE 0 = full(64 tok), 1 = half(32 tok), 2 = quarter(16 tok)
template <int NTOK, int MODE>
__global__ __launch_bounds__(256) void swmlp_win(
    const float* __restrict__ x,
    const float* __restrict__ lng, const float* __restrict__ lnb,
    const float* __restrict__ b1,
    const float* __restrict__ sg,  const float* __restrict__ sb,
    const float* __restrict__ bs,  const float* __restrict__ b2,
    const __bf16* __restrict__ w1t, const __bf16* __restrict__ w2t,
    const __bf16* __restrict__ wsb,
    float* __restrict__ out) {
  constexpr int TQ    = 256 / NTOK;           // threads per token (LN phases)
  constexpr int CH    = NTOK;                 // channels per thread (=256/TQ)
  constexpr int MT    = NTOK / 16;            // M tiles
  constexpr int NTOKP = (NTOK < 32) ? 32 : NTOK;
  constexpr int AR    = 264;                  // padded row stride (bf16 elems)
  constexpr int VTR   = NTOKP + 8;            // padded VnT row stride
  constexpr int WSROW = (MODE == 0) ? 64 : 32;

  __shared__ alignas(16) char smem[NTOK * AR * 2 * 3 + 256 * VTR * 2];
  __bf16* Abuf = (__bf16*)smem;               // LN(x) -> later G = U*V'
  __bf16* Ubuf = Abuf + NTOK * AR;            // u half (also phase-A scratch)
  __bf16* Vbuf = Ubuf + NTOK * AR;            // raw gelu v
  __bf16* VnT  = Vbuf + NTOK * AR;            // LN(v), channel-major (B-operand)
  float*  Outb = (float*)Vbuf;                // f32 out stage, overlays V+VnT

  const int tid = threadIdx.x;
  const int wv = tid >> 5, lane = tid & 31;
  const int c = lane & 15, hi = lane >> 4;
  const int wid = blockIdx.x;

  int bidx, hp0, wp0, tsh;
  if (MODE == 0) {
    bidx = wid / 225; int r = wid % 225; hp0 = (r / 15) * 8; wp0 = (r % 15) * 8; tsh = 3;
  } else if (MODE == 1) {
    int g = wid / 240, r = wid % 240; bidx = r / 15; int wi = r % 15;
    if (g < 2) { hp0 = wi * 8; wp0 = 120 + g * 4; tsh = 2; }
    else       { hp0 = 120 + (g - 2) * 4; wp0 = wi * 8; tsh = 3; }
  } else {
    int g = wid >> 4; bidx = wid & 15;
    hp0 = 120 + (g >> 1) * 4; wp0 = 120 + (g & 1) * 4; tsh = 2;
  }
  const int txm = (1 << tsh) - 1;

  const int q = tid / NTOK, tok = tid % NTOK;
  const int th = (hp0 + (tok >> tsh) + 4) & 127;
  const int tw = (wp0 + (tok & txm) + 4) & 127;
  const size_t pixbase = ((size_t)bidx * 256) * 16384 + (size_t)th * 128 + tw;

  // ---------- Phase A: gather x, LayerNorm over channels -> Abuf (bf16) ----------
  {
    float* sums = (float*)Ubuf;  // scratch: [0,256)=sum [256,512)=sumsq [512..]=mu/rs
    float s = 0.f, ss = 0.f;
#pragma unroll 4
    for (int j = 0; j < CH; ++j) {
      int d = q * CH + j;
      float v = x[pixbase + (size_t)d * 16384];
      Abuf[tok * AR + d] = (__bf16)v;
      s += v; ss += v * v;
    }
    sums[q * NTOK + tok] = s;
    sums[256 + q * NTOK + tok] = ss;
    __syncthreads();
    if (q == 0) {
      float S = 0.f, SS = 0.f;
      for (int qq = 0; qq < TQ; ++qq) { S += sums[qq * NTOK + tok]; SS += sums[256 + qq * NTOK + tok]; }
      float mu = S * (1.f / 256.f);
      float var = SS * (1.f / 256.f) - mu * mu;
      sums[512 + tok] = mu;
      sums[512 + NTOK + tok] = rsqrtf(var + 1e-5f);
    }
    __syncthreads();
    float mu = sums[512 + tok], rs = sums[512 + NTOK + tok];
    for (int j = 0; j < CH; ++j) {
      int d = q * CH + j;
      float v = (float)Abuf[tok * AR + d];
      Abuf[tok * AR + d] = (__bf16)((v - mu) * rs * lng[d] + lnb[d]);
    }
  }
  __syncthreads();

  // ---------- GEMM1: Z = A @ W1 + b1, GELU, split into U / Vraw ----------
  {
#pragma unroll
    for (int nn = 0; nn < 4; ++nn) {
      const int nt = wv * 4 + nn;                               // 32 N-tiles over 512
      const __bf16* brow = w1t + (size_t)(nt * 16 + c) * 256;
      __builtin_prefetch(w1t + (size_t)((nt + 1) * 16 + c) * 256, 0, 1);
      v8f acc[MT] = {};
#pragma unroll
      for (int kt = 0; kt < 8; ++kt) {
        v16bf bf = load_b_frag(brow, kt * 32, hi);
#pragma unroll
        for (int mt = 0; mt < MT; ++mt) {
          v16bf af = load_a_frag(Abuf, AR, mt * 16 + c, kt * 32, hi);
          acc[mt] = wmma_bf16(af, bf, acc[mt]);
        }
      }
      const float bv = b1[nt * 16 + c];
      __bf16* dst = (nt < 16) ? Ubuf : Vbuf;
      const int ch = (nt < 16) ? nt * 16 + c : nt * 16 + c - 256;
#pragma unroll
      for (int mt = 0; mt < MT; ++mt) {
#pragma unroll
        for (int e = 0; e < 8; ++e) {
          int m = mt * 16 + e + 8 * hi;
          dst[m * AR + ch] = (__bf16)gelu_f(acc[mt][e] + bv);
        }
      }
    }
  }
  __syncthreads();

  // ---------- LN(v) -> VnT (channel-major, K zero-padded for quarter) ----------
  {
    float* sums = (float*)VnT;  // scratch lives where VnT will be written
    float s = 0.f, ss = 0.f;
    for (int j = 0; j < CH; ++j) {
      int d = q * CH + j;
      float v = (float)Vbuf[tok * AR + d];
      s += v; ss += v * v;
    }
    sums[q * NTOK + tok] = s;
    sums[256 + q * NTOK + tok] = ss;
    __syncthreads();
    if (q == 0) {
      float S = 0.f, SS = 0.f;
      for (int qq = 0; qq < TQ; ++qq) { S += sums[qq * NTOK + tok]; SS += sums[256 + qq * NTOK + tok]; }
      float mu = S * (1.f / 256.f);
      float var = SS * (1.f / 256.f) - mu * mu;
      sums[512 + tok] = mu;
      sums[512 + NTOK + tok] = rsqrtf(var + 1e-5f);
    }
    __syncthreads();
    const float mu = sums[512 + tok], rs = sums[512 + NTOK + tok];
    __syncthreads();  // everyone holds mu/rs in regs; safe to overwrite VnT now
    for (int j = 0; j < CH; ++j) {
      int d = q * CH + j;
      float v = (float)Vbuf[tok * AR + d];
      VnT[d * VTR + tok] = (__bf16)((v - mu) * rs * sg[d] + sb[d]);
    }
    if (NTOK < NTOKP) {
      for (int idx = tid; idx < 256 * (NTOKP - NTOK); idx += 256) {
        int chz = idx / (NTOKP - NTOK);
        int kk = NTOK + idx % (NTOKP - NTOK);
        VnT[chz * VTR + kk] = (__bf16)0.f;
      }
    }
  }
  __syncthreads();

  // ---------- Spatial gating: V' = Ws @ Vn + bs ; G = U * V' -> Abuf ----------
  {
#pragma unroll
    for (int nn = 0; nn < 2; ++nn) {
      const int nt = wv * 2 + nn;                 // 16 N-tiles over 256 channels
      v8f acc[MT] = {};
#pragma unroll
      for (int kt = 0; kt < NTOKP / 32; ++kt) {
        v16bf bf = load_b_frag(VnT + (size_t)(nt * 16 + c) * VTR, kt * 32, hi);
#pragma unroll
        for (int mt = 0; mt < MT; ++mt) {
          v16bf af = load_a_frag(wsb, WSROW, mt * 16 + c, kt * 32, hi);
          acc[mt] = wmma_bf16(af, bf, acc[mt]);
        }
      }
      const int ch = nt * 16 + c;
#pragma unroll
      for (int mt = 0; mt < MT; ++mt) {
#pragma unroll
        for (int e = 0; e < 8; ++e) {
          int m = mt * 16 + e + 8 * hi;
          float vp = acc[mt][e] + bs[m];
          Abuf[m * AR + ch] = (__bf16)((float)Ubuf[m * AR + ch] * vp);
        }
      }
    }
  }
  __syncthreads();

  // ---------- GEMM2: Out = G @ W2 (f32 staged to LDS) ----------
  {
#pragma unroll
    for (int nn = 0; nn < 2; ++nn) {
      const int nt = wv * 2 + nn;
      const __bf16* brow = w2t + (size_t)(nt * 16 + c) * 256;
      v8f acc[MT] = {};
#pragma unroll
      for (int kt = 0; kt < 8; ++kt) {
        v16bf bf = load_b_frag(brow, kt * 32, hi);
#pragma unroll
        for (int mt = 0; mt < MT; ++mt) {
          v16bf af = load_a_frag(Abuf, AR, mt * 16 + c, kt * 32, hi);
          acc[mt] = wmma_bf16(af, bf, acc[mt]);
        }
      }
#pragma unroll
      for (int mt = 0; mt < MT; ++mt) {
#pragma unroll
        for (int e = 0; e < 8; ++e) {
          int m = mt * 16 + e + 8 * hi;
          Outb[m * 257 + nt * 16 + c] = acc[mt][e];
        }
      }
    }
  }
  __syncthreads();

  // ---------- Epilogue: + b2 + residual, coalesced scatter to NCHW ----------
  {
    const float* xb = x + pixbase;
    float* ob = out + pixbase;
    for (int j = 0; j < CH; ++j) {
      int d = q * CH + j;
      ob[(size_t)d * 16384] = Outb[tok * 257 + d] + b2[d] + xb[(size_t)d * 16384];
    }
  }
}

extern "C" void kernel_launch(void* const* d_in, const int* in_sizes, int n_in,
                              void* d_out, int out_size, void* d_ws, size_t ws_size,
                              hipStream_t stream) {
  const float* x   = (const float*)d_in[0];
  const float* lng = (const float*)d_in[1];
  const float* lnb = (const float*)d_in[2];
  const float* W1  = (const float*)d_in[3];
  const float* b1  = (const float*)d_in[4];
  const float* sg  = (const float*)d_in[5];
  const float* sb  = (const float*)d_in[6];
  const float* WsF = (const float*)d_in[7];
  const float* bsF = (const float*)d_in[8];
  const float* WsH = (const float*)d_in[9];
  const float* bsH = (const float*)d_in[10];
  const float* WsQ = (const float*)d_in[11];
  const float* bsQ = (const float*)d_in[12];
  const float* W2  = (const float*)d_in[13];
  const float* b2  = (const float*)d_in[14];
  float* out = (float*)d_out;
  __bf16* ws = (__bf16*)d_ws;

  swmlp_prep<<<(TOTAL_PACK + 255) / 256, 256, 0, stream>>>(W1, W2, WsF, WsH, WsQ, ws);

  // full windows: 16 * 15 * 15 blocks, 64 tokens each
  swmlp_win<64, 0><<<3600, 256, 0, stream>>>(
      x, lng, lnb, b1, sg, sb, bsF, b2,
      ws + OFF_W1T, ws + OFF_W2T, ws + OFF_WSF, out);
  // half windows: 4 * 16 * 15 blocks, 32 tokens each
  swmlp_win<32, 1><<<960, 256, 0, stream>>>(
      x, lng + 256, lnb + 256, b1 + 512, sg + 256, sb + 256, bsH, b2 + 256,
      ws + OFF_W1T + 131072, ws + OFF_W2T + 65536, ws + OFF_WSH, out);
  // quarter windows: 4 * 16 blocks, 16 tokens each
  swmlp_win<16, 2><<<64, 256, 0, stream>>>(
      x, lng + 512, lnb + 512, b1 + 1024, sg + 512, sb + 512, bsQ, b2 + 512,
      ws + OFF_W1T + 262144, ws + OFF_W2T + 131072, ws + OFF_WSQ, out);
}
